// SpikingNeuralSubstrate_65524021067812
// MI455X (gfx1250) — matrix operand order repaired and checked by reading, not compiled, not run on hardware
//
#include <hip/hip_runtime.h>

typedef __attribute__((ext_vector_type(16))) _Float16 v16h;
typedef __attribute__((ext_vector_type(8)))  float    v8f;

#define TSTEPS 64
#define BATCH  256
#define NIN    1024
#define NH     4096
#define NOUT   1024
#define ALPHA  0.05f   // DT/TAU = 1/20
#define THRV   1.0f

// ---------------- WMMA fragment helpers (layouts per CDNA5 ISA 7.12.2) ---

// A: 16x32 f16 (MxK). lane<16: row M=lane, halves = K {0..7, 16..23};
// lane>=16: row M=lane-16, halves = K {8..15, 24..31}.
__device__ __forceinline__ v16h load_a_frag(const _Float16* __restrict__ S, int ldk,
                                            int row, int k, int lane) {
    const int b1 = (lane & 16) ? 8 : 0;
    const _Float16* p = S + (size_t)row * ldk + k + b1;
    v16h a;
    ((uint4*)&a)[0] = *(const uint4*)(p);        // K = k+b1 .. k+b1+7
    ((uint4*)&a)[1] = *(const uint4*)(p + 16);   // K = k+b1+16 .. k+b1+23
    return a;
}

// B: 32x16 f16 (KxN). lane<16: col N=lane, K {0..7, 16..23}; lane>=16: col N=lane-16, K {8..15, 24..31}.
// Weights stored [N, K] row-major -> each lane reads two 16-byte runs, 32 bytes apart.
__device__ __forceinline__ v16h load_b_frag(const _Float16* __restrict__ W, int ldk,
                                            int col, int k, int lane) {
    const int kb = (lane & 16) ? 8 : 0;
    const _Float16* p = W + (size_t)col * ldk + k + kb;
    v16h b;
    ((uint4*)&b)[0] = *(const uint4*)(p);
    ((uint4*)&b)[1] = *(const uint4*)(p + 16);
    return b;
}

__device__ __forceinline__ v8f wmma_f16(v16h a, v16h b, v8f c) {
    return __builtin_amdgcn_wmma_f32_16x16x32_f16(false, a, false, b, (short)0, c,
                                                  false, false);
}

// ---------------- GEMM + fused LIF: hidden layer ------------------------------------
// I_h = s_in @ W_ih^T + s_h @ W_hh^T ; then v_h LIF update -> s_h_next (f16)
// Per-wave tile: 32(M) x 64(N) -> 8 accumulators, B fragments reused across 2 M rows.
// Waves: (BATCH/32) x (NH/64) = 8 x 64 = 512 -> 64 blocks x 8 waves.
// All 8 waves of a block share tile_n (same weight columns) for WGP$ reuse.
__global__ __launch_bounds__(256) void snn_gemm_hidden(
    const _Float16* __restrict__ s_in,   // [BATCH, NIN]
    const _Float16* __restrict__ s_h,    // [BATCH, NH]
    const _Float16* __restrict__ W_ih,   // [NH, NIN]
    const _Float16* __restrict__ W_hh,   // [NH, NH]
    float* __restrict__ v_h,             // [BATCH, NH]
    _Float16* __restrict__ s_h_next)     // [BATCH, NH]
{
    const int lane   = threadIdx.x & 31;
    const int wave   = (blockIdx.x * blockDim.x + threadIdx.x) >> 5;
    const int tile_m = wave & 7;         // 8 tiles of M=32 over BATCH=256
    const int tile_n = wave >> 3;        // 64 tiles of N=64 over NH=4096
    const int row0   = tile_m << 5;
    const int col0   = tile_n << 6;
    const int arow0  = row0 + (lane & 15);
    const int bcol   = lane & 15;

    v8f acc[2][4];
    #pragma unroll
    for (int h = 0; h < 2; ++h)
        #pragma unroll
        for (int j = 0; j < 4; ++j) acc[h][j] = v8f{0.f};

    for (int k = 0; k < NIN; k += 32) {
        v16h a0 = load_a_frag(s_in, NIN, arow0,      k, lane);
        v16h a1 = load_a_frag(s_in, NIN, arow0 + 16, k, lane);
        #pragma unroll
        for (int j = 0; j < 4; ++j) {
            v16h b = load_b_frag(W_ih, NIN, col0 + (j << 4) + bcol, k, lane);
            acc[0][j] = wmma_f16(a0, b, acc[0][j]);
            acc[1][j] = wmma_f16(a1, b, acc[1][j]);
        }
    }
    for (int k = 0; k < NH; k += 32) {
        v16h a0 = load_a_frag(s_h, NH, arow0,      k, lane);
        v16h a1 = load_a_frag(s_h, NH, arow0 + 16, k, lane);
        #pragma unroll
        for (int j = 0; j < 4; ++j) {
            v16h b = load_b_frag(W_hh, NH, col0 + (j << 4) + bcol, k, lane);
            acc[0][j] = wmma_f16(a0, b, acc[0][j]);
            acc[1][j] = wmma_f16(a1, b, acc[1][j]);
        }
    }

    // C/D layout: VGPR i -> row i (lanes 0-15) / i+8 (lanes 16-31), col = lane&15
    const int nn = lane & 15;
    #pragma unroll
    for (int h = 0; h < 2; ++h) {
        const int mb = row0 + (h << 4) + ((lane & 16) ? 8 : 0);
        #pragma unroll
        for (int j = 0; j < 4; ++j) {
            const int col = col0 + (j << 4) + nn;
            #pragma unroll
            for (int i = 0; i < 8; ++i) {
                const size_t idx = (size_t)(mb + i) * NH + col;
                float v = v_h[idx];
                v += ALPHA * (acc[h][j][i] - v);
                const float s = (v >= THRV) ? 1.0f : 0.0f;
                v_h[idx] = v * (1.0f - s);
                s_h_next[idx] = (_Float16)s;
            }
        }
    }
}

// ---------------- GEMM + fused LIF: output layer ------------------------------------
// I_out = s_h @ W_ho^T ; v_out LIF -> spikes written as f32 into d_out[t]
// Small GEMM: keep 16x64 per-wave tile for parallelism. 256 waves -> 32 blocks.
__global__ __launch_bounds__(256) void snn_gemm_out(
    const _Float16* __restrict__ s_h,    // [BATCH, NH]
    const _Float16* __restrict__ W_ho,   // [NOUT, NH]
    float* __restrict__ v_out,           // [BATCH, NOUT]
    float* __restrict__ out_t)           // [BATCH, NOUT]
{
    const int lane   = threadIdx.x & 31;
    const int wave   = (blockIdx.x * blockDim.x + threadIdx.x) >> 5;
    const int tile_m = wave & 15;        // 16 tiles of M=16
    const int tile_n = wave >> 4;        // 16 tiles of N=64
    const int row0   = tile_m << 4;
    const int col0   = tile_n << 6;
    const int arow   = row0 + (lane & 15);
    const int bcol   = lane & 15;

    v8f acc[4] = {v8f{0.f}, v8f{0.f}, v8f{0.f}, v8f{0.f}};

    for (int k = 0; k < NH; k += 32) {
        v16h a = load_a_frag(s_h, NH, arow, k, lane);
        #pragma unroll
        for (int j = 0; j < 4; ++j) {
            v16h b = load_b_frag(W_ho, NH, col0 + (j << 4) + bcol, k, lane);
            acc[j] = wmma_f16(a, b, acc[j]);
        }
    }

    const int mb = row0 + ((lane & 16) ? 8 : 0);
    const int nn = lane & 15;
    #pragma unroll
    for (int j = 0; j < 4; ++j) {
        const int col = col0 + (j << 4) + nn;
        #pragma unroll
        for (int i = 0; i < 8; ++i) {
            const size_t idx = (size_t)(mb + i) * NOUT + col;
            float v = v_out[idx];
            v += ALPHA * (acc[j][i] - v);
            const float s = (v >= THRV) ? 1.0f : 0.0f;
            v_out[idx] = v * (1.0f - s);
            out_t[idx] = s;
        }
    }
}

// ---------------- Elementwise kernels -----------------------------------------------

// Input-group LIF: I_in = x_t (external current), produces s_in_next as f16
__global__ void snn_lif_input(const float* __restrict__ x_t,
                              float* __restrict__ v_in,
                              _Float16* __restrict__ s_in_next, int n) {
    const int i = blockIdx.x * blockDim.x + threadIdx.x;
    if (i < n) {
        float v = v_in[i];
        v += ALPHA * (x_t[i] - v);
        const float s = (v >= THRV) ? 1.0f : 0.0f;
        v_in[i] = v * (1.0f - s);
        s_in_next[i] = (_Float16)s;
    }
}

__global__ void snn_f32_to_f16(const float* __restrict__ src,
                               _Float16* __restrict__ dst, int n) {
    const int i = blockIdx.x * blockDim.x + threadIdx.x;
    if (i < n) dst[i] = (_Float16)src[i];
}

__global__ void snn_zero_u32(unsigned int* __restrict__ p, int n) {
    const int i = blockIdx.x * blockDim.x + threadIdx.x;
    if (i < n) p[i] = 0u;
}

// ---------------- Host driver --------------------------------------------------------

extern "C" void kernel_launch(void* const* d_in, const int* in_sizes, int n_in,
                              void* d_out, int out_size, void* d_ws, size_t ws_size,
                              hipStream_t stream) {
    const float* x_ext = (const float*)d_in[0];  // [T, B, NIN]
    const float* W_ih  = (const float*)d_in[1];  // [NH, NIN]
    const float* W_hh  = (const float*)d_in[2];  // [NH, NH]
    const float* W_ho  = (const float*)d_in[3];  // [NOUT, NH]
    float* out = (float*)d_out;                  // [T, B, NOUT]

    char* ws = (char*)d_ws;
    size_t off = 0;
    _Float16* W_ih_h = (_Float16*)(ws + off); off += (size_t)NH * NIN * 2;
    _Float16* W_hh_h = (_Float16*)(ws + off); off += (size_t)NH * NH  * 2;
    _Float16* W_ho_h = (_Float16*)(ws + off); off += (size_t)NOUT * NH * 2;

    // state region (zeroed each call, contiguous)
    char* state = ws + off;
    size_t soff = off;
    float*    v_in  = (float*)(ws + soff);    soff += (size_t)BATCH * NIN  * 4;
    float*    v_h   = (float*)(ws + soff);    soff += (size_t)BATCH * NH   * 4;
    float*    v_out = (float*)(ws + soff);    soff += (size_t)BATCH * NOUT * 4;
    _Float16* s_in[2];
    _Float16* s_h[2];
    s_in[0] = (_Float16*)(ws + soff); soff += (size_t)BATCH * NIN * 2;
    s_in[1] = (_Float16*)(ws + soff); soff += (size_t)BATCH * NIN * 2;
    s_h[0]  = (_Float16*)(ws + soff); soff += (size_t)BATCH * NH  * 2;
    s_h[1]  = (_Float16*)(ws + soff); soff += (size_t)BATCH * NH  * 2;
    const size_t state_bytes = soff - off;
    (void)state; (void)ws_size; (void)n_in; (void)in_sizes; (void)out_size;

    // 1) convert weights to f16 (spikes are exactly {0,1} in f16; f32 accumulate)
    {
        int n1 = NH * NIN, n2 = NH * NH, n3 = NOUT * NH;
        snn_f32_to_f16<<<(n1 + 255) / 256, 256, 0, stream>>>(W_ih, W_ih_h, n1);
        snn_f32_to_f16<<<(n2 + 255) / 256, 256, 0, stream>>>(W_hh, W_hh_h, n2);
        snn_f32_to_f16<<<(n3 + 255) / 256, 256, 0, stream>>>(W_ho, W_ho_h, n3);
    }
    // 2) zero all membrane potentials + spike buffers (determinism across calls)
    {
        int ndw = (int)(state_bytes / 4);
        snn_zero_u32<<<(ndw + 255) / 256, 256, 0, stream>>>((unsigned int*)state, ndw);
    }

    // 3) time loop: currents from PREVIOUS spikes (double-buffered)
    int cur = 0;
    for (int t = 0; t < TSTEPS; ++t) {
        const int nxt = cur ^ 1;
        snn_lif_input<<<(BATCH * NIN + 255) / 256, 256, 0, stream>>>(
            x_ext + (size_t)t * BATCH * NIN, v_in, s_in[nxt], BATCH * NIN);
        snn_gemm_hidden<<<64, 256, 0, stream>>>(
            s_in[cur], s_h[cur], W_ih_h, W_hh_h, v_h, s_h[nxt]);
        snn_gemm_out<<<32, 256, 0, stream>>>(
            s_h[cur], W_ho_h, v_out, out + (size_t)t * BATCH * NOUT);
        cur = nxt;
    }
}
